// ClothesBasedAdversarialLossWithMemoryBank_10307921511291
// MI455X (gfx1250) — compile-verified, matrix operand order checked by reading, not run on hardware
//
#include <hip/hip_runtime.h>
#include <hip/hip_bf16.h>

#define MOMENTUM 0.2f
#define SCALEF   16.0f
#define EPS_LS   0.1f
#define NUMC     100000
#define FEAT     512
#define BATCH    512
#define MAXPOS   (1 << 20)

typedef __attribute__((ext_vector_type(16))) __bf16 v16bf;
typedef __attribute__((ext_vector_type(8)))  __bf16 v8bf;
typedef __attribute__((ext_vector_type(8)))  float  v8f;

union BfU { v16bf v; v8bf h[2]; };

__device__ __forceinline__ float wave_sum32(float v) {
#pragma unroll
  for (int off = 1; off < 32; off <<= 1) v += __shfl_xor(v, off, 32);
  return v;
}

// ---------------- init scratch ----------------
__global__ void k_init(int* slot_map, float* seg_sum, int* slot_cnt,
                       float* S_total, float* S_pos, float* Psum,
                       int* nslots, unsigned* pos_cnt, float* loss) {
  int i = blockIdx.x * blockDim.x + threadIdx.x;
  if (i < NUMC) slot_map[i] = -1;
  if (i < BATCH * FEAT) seg_sum[i] = 0.f;
  if (i < BATCH) { slot_cnt[i] = 0; S_total[i] = 0.f; S_pos[i] = 0.f; Psum[i] = 0.f; }
  if (i == 0) { *nslots = 0; *pos_cnt = 0u; *loss = 0.f; }
}

// ---------------- compact slot assignment for touched classes ----------------
__global__ void k_slots(const int* targets, int* slot_map, int* nslots) {
  int i = threadIdx.x;            // 1 block of BATCH threads
  int c = targets[i];
  int prev = atomicCAS(&slot_map[c], -1, -2);
  if (prev == -1) {
    int s = atomicAdd(nslots, 1);
    slot_map[c] = s;
  }
}

// ---------------- segment sum into compact slots ----------------
__global__ void k_seg(const float* inputs, const int* targets, const int* slot_map,
                      float* seg_sum, int* slot_cnt) {
  int i = blockIdx.x;             // sample
  int s = slot_map[targets[i]];
  if (threadIdx.x == 0) atomicAdd(&slot_cnt[s], 1);
  for (int d = threadIdx.x; d < FEAT; d += blockDim.x)
    atomicAdd(&seg_sum[s * FEAT + d], inputs[(size_t)i * FEAT + d]);
}

// ---------------- blend + L2 normalize memory -> bf16 ----------------
__global__ void k_memnorm(const float* mem, const int* slot_map, const float* seg_sum,
                          const int* slot_cnt, __bf16* memB) {
  int c = blockIdx.x;             // one class per 128-thread block
  int t = threadIdx.x;
  int s = slot_map[c];
  float cntf = 1.f;
  if (s >= 0) cntf = (float)slot_cnt[s];
  const float* mrow = mem + (size_t)c * FEAT;
  float vals[4];
  float ss = 0.f;
#pragma unroll
  for (int j = 0; j < 4; j++) {
    int d = t + j * 128;
    float v = mrow[d];
    if (s >= 0) v = MOMENTUM * v + (1.f - MOMENTUM) * (seg_sum[s * FEAT + d] / cntf);
    vals[j] = v;
    ss += v * v;
  }
  ss = wave_sum32(ss);
  __shared__ float red[4];
  if ((t & 31) == 0) red[t >> 5] = ss;
  __syncthreads();
  float tot = red[0] + red[1] + red[2] + red[3];
  float inv = 1.f / fmaxf(sqrtf(tot), 1e-12f);
#pragma unroll
  for (int j = 0; j < 4; j++)
    memB[(size_t)c * FEAT + t + j * 128] = (__bf16)(vals[j] * inv);
}

// ---------------- L2 normalize inputs -> bf16 ----------------
__global__ void k_innorm(const float* inputs, __bf16* inB) {
  int b = blockIdx.x;
  int t = threadIdx.x;
  const float* row = inputs + (size_t)b * FEAT;
  float vals[4];
  float ss = 0.f;
#pragma unroll
  for (int j = 0; j < 4; j++) {
    int d = t + j * 128;
    float v = row[d];
    vals[j] = v;
    ss += v * v;
  }
  ss = wave_sum32(ss);
  __shared__ float red[4];
  if ((t & 31) == 0) red[t >> 5] = ss;
  __syncthreads();
  float tot = red[0] + red[1] + red[2] + red[3];
  float inv = 1.f / fmaxf(sqrtf(tot), 1e-12f);
#pragma unroll
  for (int j = 0; j < 4; j++)
    inB[(size_t)b * FEAT + t + j * 128] = (__bf16)(vals[j] * inv);
}

// ---------------- main WMMA GEMM + streaming softmax stats ----------------
// block = 256 threads = 8 waves; wave = 16(batch rows) x 16(classes) tile
// grid.x = 16 row blocks (32 rows each), grid.y = 64 column strips
__global__ __launch_bounds__(256, 1)
void k_gemm(const __bf16* __restrict__ inB, const __bf16* __restrict__ memB,
            const float* __restrict__ pmask,
            float* S_total, float* S_pos, float* Psum,
            unsigned* pos_cnt, int* pos_b, int* pos_c, float* pos_s) {
  const int lane = threadIdx.x & 31;
  const int wave = threadIdx.x >> 5;
  const int mbase = blockIdx.x * 32 + (wave >> 2) * 16;
  const int ncsub = wave & 3;
  const int lo = lane & 15;
  const int hi = lane >> 4;          // 0 or 1
  const int koff = hi * 8;           // K sub-offset per ISA bf16 A/B layout

  // Preload all A fragments for this wave's 16 rows (full K = 512)
  v16bf afrag[16];
  const __bf16* arow = inB + (size_t)(mbase + lo) * FEAT;
#pragma unroll
  for (int kc = 0; kc < 16; kc++) {
    BfU u;
    u.h[0] = *(const v8bf*)(arow + kc * 32 + koff);        // elems 0..7  : K = kc*32 + koff + i
    u.h[1] = *(const v8bf*)(arow + kc * 32 + 16 + koff);   // elems 8..15 : K = kc*32 + 16 + koff + i
    afrag[kc] = u.v;
  }

  float accT[8], accP[8], accM[8];
#pragma unroll
  for (int r = 0; r < 8; r++) { accT[r] = 0.f; accP[r] = 0.f; accM[r] = 0.f; }

  for (int colbase = blockIdx.y * 64; colbase < NUMC; colbase += gridDim.y * 64) {
    int nbase = colbase + ncsub * 16;
    if (nbase >= NUMC) continue;                           // NUMC % 16 == 0: no partial tiles
    const __bf16* brow = memB + (size_t)(nbase + lo) * FEAT;
    v8f acc = {0.f, 0.f, 0.f, 0.f, 0.f, 0.f, 0.f, 0.f};
#pragma unroll
    for (int kc = 0; kc < 16; kc++) {
      BfU u;
      u.h[0] = *(const v8bf*)(brow + kc * 32 + koff);
      u.h[1] = *(const v8bf*)(brow + kc * 32 + 16 + koff);
      acc = __builtin_amdgcn_wmma_f32_16x16x32_bf16(false, afrag[kc], false, u.v,
                                                    (short)0, acc, false, false);
    }
    int cg = nbase + lo;                                   // class column (N = lane&15)
#pragma unroll
    for (int r = 0; r < 8; r++) {                          // row M = r + 8*hi
      int b = mbase + r + 8 * hi;
      float sim = acc[r] * SCALEF;
      float pm = pmask[(size_t)b * NUMC + cg];
      float e = __expf(sim);
      accT[r] += e;
      accP[r] += e * pm;
      accM[r] += pm;
      if (pm > 0.f) {
        unsigned idx = atomicAdd(pos_cnt, 1u);
        if (idx < MAXPOS) { pos_b[idx] = b; pos_c[idx] = cg; pos_s[idx] = sim; }
      }
    }
  }

  // reduce over the 16 lanes of each half (they share the same output row)
#pragma unroll
  for (int r = 0; r < 8; r++) {
    float t = accT[r], p = accP[r], m = accM[r];
#pragma unroll
    for (int off = 1; off < 16; off <<= 1) {
      t += __shfl_xor(t, off, 32);
      p += __shfl_xor(p, off, 32);
      m += __shfl_xor(m, off, 32);
    }
    if (lo == 0) {
      int b = mbase + r + 8 * hi;
      atomicAdd(&S_total[b], t);
      atomicAdd(&S_pos[b], p);
      atomicAdd(&Psum[b], m);
    }
  }
}

// ---------------- loss over the sparse positive list ----------------
__global__ void k_final(const int* targets, const float* S_total, const float* S_pos,
                        const float* Psum, const unsigned* pos_cnt,
                        const int* pos_b, const int* pos_c, const float* pos_s,
                        float* loss) {
  unsigned n = *pos_cnt;
  if (n > MAXPOS) n = MAXPOS;
  float local = 0.f;
  for (unsigned i = blockIdx.x * blockDim.x + threadIdx.x; i < n;
       i += gridDim.x * blockDim.x) {
    int b = pos_b[i];
    int c = pos_c[i];
    float sim = pos_s[i];
    float e = __expf(sim);
    float Sneg = S_total[b] - S_pos[b];                    // sum over negatives
    float lp = sim - logf(Sneg + e);
    float w = EPS_LS / Psum[b];                            // pm == 1 at listed entries
    if (c == targets[b]) w += (1.f - EPS_LS);
    local += -w * lp;
  }
  local = wave_sum32(local);
  __shared__ float red[8];
  if ((threadIdx.x & 31) == 0) red[threadIdx.x >> 5] = local;
  __syncthreads();
  if (threadIdx.x == 0) {
    float s = 0.f;
#pragma unroll
    for (int i = 0; i < 8; i++) s += red[i];
    atomicAdd(loss, s);
  }
}

__global__ void k_out(const float* loss, float* out) {
  out[0] = loss[0] * (1.0f / (float)BATCH);
}

extern "C" void kernel_launch(void* const* d_in, const int* in_sizes, int n_in,
                              void* d_out, int out_size, void* d_ws, size_t ws_size,
                              hipStream_t stream) {
  (void)in_sizes; (void)n_in; (void)out_size; (void)ws_size;
  const float* inputs  = (const float*)d_in[0];
  const float* pmask   = (const float*)d_in[1];
  const float* mem     = (const float*)d_in[2];
  const int*   targets = (const int*)d_in[3];
  float* out = (float*)d_out;

  char* base = (char*)d_ws;
  size_t off = 0;
  auto take = [&](size_t bytes) -> void* {
    void* p = base + off;
    off = (off + bytes + 255) & ~(size_t)255;
    return p;
  };
  __bf16*   memB     = (__bf16*)take((size_t)NUMC * FEAT * 2);
  __bf16*   inB      = (__bf16*)take((size_t)BATCH * FEAT * 2);
  int*      slot_map = (int*)take((size_t)NUMC * 4);
  float*    seg_sum  = (float*)take((size_t)BATCH * FEAT * 4);
  int*      slot_cnt = (int*)take((size_t)BATCH * 4);
  float*    S_total  = (float*)take((size_t)BATCH * 4);
  float*    S_pos    = (float*)take((size_t)BATCH * 4);
  float*    Psum     = (float*)take((size_t)BATCH * 4);
  int*      nslots   = (int*)take(4);
  unsigned* pos_cnt  = (unsigned*)take(4);
  float*    loss     = (float*)take(4);
  int*      pos_b    = (int*)take((size_t)MAXPOS * 4);
  int*      pos_c    = (int*)take((size_t)MAXPOS * 4);
  float*    pos_s    = (float*)take((size_t)MAXPOS * 4);

  k_init<<<(BATCH * FEAT + 255) / 256, 256, 0, stream>>>(
      slot_map, seg_sum, slot_cnt, S_total, S_pos, Psum, nslots, pos_cnt, loss);
  k_slots<<<1, BATCH, 0, stream>>>(targets, slot_map, nslots);
  k_seg<<<BATCH, 256, 0, stream>>>(inputs, targets, slot_map, seg_sum, slot_cnt);
  k_memnorm<<<NUMC, 128, 0, stream>>>(mem, slot_map, seg_sum, slot_cnt, memB);
  k_innorm<<<BATCH, 128, 0, stream>>>(inputs, inB);
  k_gemm<<<dim3(16, 64), 256, 0, stream>>>(inB, memB, pmask, S_total, S_pos, Psum,
                                           pos_cnt, pos_b, pos_c, pos_s);
  k_final<<<64, 256, 0, stream>>>(targets, S_total, S_pos, Psum, pos_cnt,
                                  pos_b, pos_c, pos_s, loss);
  k_out<<<1, 1, 0, stream>>>(loss, out);
}
